// siameseRPN_4483945857221
// MI455X (gfx1250) — matrix operand order repaired and checked by reading, not compiled, not run on hardware
//
#include <hip/hip_runtime.h>

// ---------------------------------------------------------------------------
// Siamese-RPN head for MI455X (gfx1250, wave32, WMMA + TDM).
// bf16 data / f32 accumulate via v_wmma_f32_16x16x32_bf16; the dominant 7x7
// cross-correlations run as 49 shifted 1x1 GEMMs with the per-tap weight slab
// staged in LDS by the Tensor Data Mover (double-buffered, s_wait_tensorcnt).
// ---------------------------------------------------------------------------

typedef __bf16 bf16;
typedef __attribute__((ext_vector_type(16))) __bf16 v16bf;
typedef __attribute__((ext_vector_type(8)))  __bf16 v8bf;
typedef __attribute__((ext_vector_type(8)))  float  v8f;
typedef __attribute__((ext_vector_type(2)))  float  v2f;
typedef unsigned int __attribute__((ext_vector_type(4))) u32x4;
typedef int          __attribute__((ext_vector_type(4))) i32x4;
typedef int          __attribute__((ext_vector_type(8))) i32x8;

#define DIN    512
#define CORRC  256
#define NANC   9
#define HH     64
#define HWPIX  4096
#define NTPL   8
#define QCOL   392     // 8 * 49 template pixels
#define MKROW  13824   // 4608 + 9216 kernel-gen rows
#define F_CLS  144     // 8 * 18 correlation filters (cls)
#define F_BOX  288     // 8 * 36 correlation filters (box)
#define PRE_N  2000
#define POST_N 300
#define NEGINF (-3.402823e38f)

#define CORR_MT 48     // filters per block (3 x 16)
#define CORR_NT 64     // pixels per block  (4 waves x 16)

#if __has_builtin(__builtin_amdgcn_tensor_load_to_lds) && \
    __has_builtin(__builtin_amdgcn_s_wait_tensorcnt)
#define HAVE_TDM 1
#else
#define HAVE_TDM 0
#endif
#if __has_include(<hip/amd_detail/amd_gfx1250_TDM.h>)
#define TDM_SIX_ARGS 1   // therock-10.0 headers -> clang-23 -> 6-arg builtin
#else
#define TDM_SIX_ARGS 0   // ROCm 7.2 clang-22 -> 5-arg builtin
#endif

// Compile-time path report (shows up in stderr even on successful compiles).
#if HAVE_TDM
#if TDM_SIX_ARGS
#warning "CDNA5 path: tensor_load_to_lds ENABLED (6-arg form) + s_wait_tensorcnt"
#else
#warning "CDNA5 path: tensor_load_to_lds ENABLED (5-arg form) + s_wait_tensorcnt"
#endif
#else
#warning "CDNA5 path: TDM builtins NOT available -> cooperative-copy fallback"
#endif

// ---------------------------------------------------------------------------
// Fragment loaders for the wave32 16-bit WMMA operands, in contiguous k order (same for A and B).
//   A (16x32, MxK): lane m=l&15, hi=l>>4.  V0..7 <- K = kk+16*hi+[0..15]
//   B (32x16, KxN): lane n=l&15, hi=l>>4.  V0..7 <- K = kk+16*hi+[0..15]
// A is [M][K], B is [N][K], so both are contiguous 16B/32B loads.
// ---------------------------------------------------------------------------
__device__ __forceinline__ v16bf load_a_frag(const bf16* row, int kk, int hi) {
    return *reinterpret_cast<const v16bf*>(row + kk + 16 * hi);
}
__device__ __forceinline__ v16bf load_b_frag(const bf16* row, int kk, int hi) {
    return *reinterpret_cast<const v16bf*>(row + kk + 16 * hi);
}

// ---------------------------------------------------------------------------
// Tensor Data Mover: DMA a [rows x 256] bf16 slab (row-major, stride 256)
// from global memory into LDS.  D# built per CDNA5 ISA 08_async_tensor §8.3/8.4:
//   group0: count=1 | lds_addr | global_addr[56:0] | type=2
//   group1: wg_mask=0, data_size=1(2B), tensor_dim0=256, tensor_dim1=rows,
//           tile_dim0=256, tile_dim1=rows, tile_dim2=0, dim0_stride=256
// ---------------------------------------------------------------------------
#if HAVE_TDM
__device__ __forceinline__ void tdm_load_slab(const bf16* gsrc, unsigned lds_off,
                                              int rows) {
    unsigned long long ga = (unsigned long long)(__SIZE_TYPE__)gsrc;
    u32x4 g0;
    g0.x = 1u;                                              // count=1 (valid user D#)
    g0.y = lds_off;                                         // lds_addr (bytes)
    g0.z = (unsigned)(ga & 0xffffffffu);                    // global_addr[31:0]
    g0.w = (unsigned)((ga >> 32) & 0x1ffffffu) | (2u << 30);// global_addr[56:32]|type=2
    i32x8 g1;
    g1[0] = (1 << 16);                 // workgroup_mask=0, data_size=1 (2 bytes)
    g1[1] = (256 & 0xffff) << 16;      // tensor_dim0 lo16 (=256); atomic_barrier=0
    g1[2] = (rows & 0xffff) << 16;     // tensor_dim0 hi16=0 | tensor_dim1 lo16
    g1[3] = (256 << 16);               // tensor_dim1 hi16=0 | tile_dim0=256
    g1[4] = (rows & 0xffff);           // tile_dim1=rows, tile_dim2=0
    g1[5] = 256;                       // tensor_dim0_stride[31:0] = 256 elems
    g1[6] = 0;                         // stride0 hi | stride1 lo
    g1[7] = 0;                         // stride1 hi
    i32x4 z4 = {0, 0, 0, 0};
#if TDM_SIX_ARGS
    i32x8 z8 = {0, 0, 0, 0, 0, 0, 0, 0};
    __builtin_amdgcn_tensor_load_to_lds(g0, g1, z4, z4, z8, 0);
#else
    __builtin_amdgcn_tensor_load_to_lds(g0, g1, z4, z4, 0);
#endif
}
#endif

// ---------------------------------------------------------------------------
// Packing kernels (f32 -> bf16, with the transposes the WMMA layout wants)
// ---------------------------------------------------------------------------
__global__ void pack_xt_kernel(const float* __restrict__ tf, bf16* __restrict__ xt) {
    int p = blockIdx.x;                       // pixel
    for (int c = threadIdx.x; c < DIN; c += 256)
        xt[p * DIN + c] = (bf16)tf[c * HWPIX + p];
}
__global__ void pack_tt_kernel(const float* __restrict__ tpl, bf16* __restrict__ tt) {
    int q = blockIdx.x;                       // n*49+s
    int n = q / 49, s = q % 49;
    for (int c = threadIdx.x; c < DIN; c += 256)
        tt[q * DIN + c] = (bf16)tpl[(n * DIN + c) * 49 + s];
}
__global__ void pack_wcc_kernel(const float* __restrict__ wc, const float* __restrict__ wb,
                                bf16* __restrict__ dst) {
    int m = blockIdx.x;                       // 0..511
    const float* src = (m < CORRC) ? (wc + m * DIN) : (wb + (m - CORRC) * DIN);
    for (int k = threadIdx.x; k < DIN; k += 256)
        dst[m * DIN + k] = (bf16)src[k];
}
__global__ void pack_wk_kernel(const float* __restrict__ wcs, const float* __restrict__ wbp,
                               bf16* __restrict__ dst) {
    int m = blockIdx.x;                       // 0..13823
    const float* src = (m < 4608) ? (wcs + m * DIN) : (wbp + (m - 4608) * DIN);
    for (int k = threadIdx.x; k < DIN; k += 256)
        dst[m * DIN + k] = (bf16)src[k];
}
__global__ void pack_bias2_kernel(const float* __restrict__ a, const float* __restrict__ b,
                                  float* __restrict__ dst, int na, int ntot) {
    int t = blockIdx.x * 256 + threadIdx.x;
    if (t < ntot) dst[t] = (t < na) ? a[t] : b[t - na];
}

// ---------------------------------------------------------------------------
// bf16 WMMA GEMM:  C[M][N] = A[M][K] * B[N][K]^T + bias[m]
// One wave per 64(M) x 16(N) tile: 4 accumulators share one B fragment per
// k-step (2.5 loads/WMMA instead of 4).  M must be a multiple of 64.
// Full N-tiles take an unguarded fast path (uniform branch) so the hot loop
// has no exec-mask juggling; only the last GEMM2 tile runs the guarded path.
// ---------------------------------------------------------------------------
__global__ void gemm_bf16_kernel(const bf16* __restrict__ A, const bf16* __restrict__ B,
                                 float* __restrict__ C, const float* __restrict__ bias,
                                 int N, int K) {
    const int n0 = blockIdx.x * 16, m0 = blockIdx.y * 64;
    const int l  = threadIdx.x, lo = l & 15, hi = l >> 4;
    const bf16* a0r = A + (size_t)(m0      + lo) * K;
    const bf16* a1r = A + (size_t)(m0 + 16 + lo) * K;
    const bf16* a2r = A + (size_t)(m0 + 32 + lo) * K;
    const bf16* a3r = A + (size_t)(m0 + 48 + lo) * K;
    const int   nn   = n0 + lo;
    const bf16* brow = B + (size_t)nn * K;
    v8f acc0 = {}, acc1 = {}, acc2 = {}, acc3 = {};
    if (n0 + 16 <= N) {                       // uniform: whole tile in-bounds
        for (int kk = 0; kk < K; kk += 32) {
            v16bf bv = load_b_frag(brow, kk, hi);
            v16bf a0 = load_a_frag(a0r, kk, hi);
            v16bf a1 = load_a_frag(a1r, kk, hi);
            v16bf a2 = load_a_frag(a2r, kk, hi);
            v16bf a3 = load_a_frag(a3r, kk, hi);
            acc0 = __builtin_amdgcn_wmma_f32_16x16x32_bf16(false, a0, false, bv,
                                                           (short)0, acc0, false, false);
            acc1 = __builtin_amdgcn_wmma_f32_16x16x32_bf16(false, a1, false, bv,
                                                           (short)0, acc1, false, false);
            acc2 = __builtin_amdgcn_wmma_f32_16x16x32_bf16(false, a2, false, bv,
                                                           (short)0, acc2, false, false);
            acc3 = __builtin_amdgcn_wmma_f32_16x16x32_bf16(false, a3, false, bv,
                                                           (short)0, acc3, false, false);
        }
    } else {                                  // ragged edge (GEMM2: N=392)
        const bool binb = (nn < N);
        v16bf bz = {};
        for (int kk = 0; kk < K; kk += 32) {
            v16bf bv = bz;
            if (binb) bv = load_b_frag(brow, kk, hi);
            v16bf a0 = load_a_frag(a0r, kk, hi);
            v16bf a1 = load_a_frag(a1r, kk, hi);
            v16bf a2 = load_a_frag(a2r, kk, hi);
            v16bf a3 = load_a_frag(a3r, kk, hi);
            acc0 = __builtin_amdgcn_wmma_f32_16x16x32_bf16(false, a0, false, bv,
                                                           (short)0, acc0, false, false);
            acc1 = __builtin_amdgcn_wmma_f32_16x16x32_bf16(false, a1, false, bv,
                                                           (short)0, acc1, false, false);
            acc2 = __builtin_amdgcn_wmma_f32_16x16x32_bf16(false, a2, false, bv,
                                                           (short)0, acc2, false, false);
            acc3 = __builtin_amdgcn_wmma_f32_16x16x32_bf16(false, a3, false, bv,
                                                           (short)0, acc3, false, false);
        }
    }
    const int ng = n0 + lo;
    if (ng < N) {
#pragma unroll
        for (int v = 0; v < 8; ++v) {
            int mg = m0 + v + 8 * hi;
            C[(size_t)(mg)      * N + ng] = acc0[v] + bias[mg];
            C[(size_t)(mg + 16) * N + ng] = acc1[v] + bias[mg + 16];
            C[(size_t)(mg + 32) * N + ng] = acc2[v] + bias[mg + 32];
            C[(size_t)(mg + 48) * N + ng] = acc3[v] + bias[mg + 48];
        }
    }
}

// ---------------------------------------------------------------------------
// 7x7 cross-correlation as 49 shifted 1x1 GEMMs.
// Block = 128 threads (4 waves) computing a 48(filter) x 64(pixel) tile.
// The 48x256 bf16 weight slab for tap s is staged in LDS by the TDM
// (double-buffered; wave 0 issues tensor_load_to_lds for tap s+1, then
// s_wait_tensorcnt 1 -> tap s has landed because TDM retires in order).
// Each wave: 3 accumulators sharing one B fragment per k-step.
// ---------------------------------------------------------------------------
__global__ void corr_wmma_kernel(const bf16* __restrict__ Amat,   // [49][F][256]
                                 const bf16* __restrict__ TNt,    // [4096][256]
                                 float* __restrict__ OUT, int F) {
    __shared__ bf16 slab[2][CORR_MT * CORRC];   // 2 x 24 KB of the 320 KB WGP LDS
    const int tid  = threadIdx.x;
    const int wave = tid >> 5;
    const int l = tid & 31, lo = l & 15, hi = l >> 4;
    const int n0 = blockIdx.x * CORR_NT + wave * 16;  // pixel base for this wave
    const int m0 = blockIdx.y * CORR_MT;              // filter base for this block
    const int pix = n0 + lo;
    const int y = pix >> 6, x = pix & 63;
    v8f acc0 = {}, acc1 = {}, acc2 = {};
    v16bf bz = {};

#if HAVE_TDM
    if (wave == 0)
        tdm_load_slab(Amat + (size_t)(0 * F + m0) * CORRC,
                      (unsigned)(__SIZE_TYPE__)&slab[0][0], CORR_MT);
#endif
    for (int s = 0; s < 49; ++s) {
#if HAVE_TDM
        if (wave == 0) {
            if (s + 1 < 49) {
                tdm_load_slab(Amat + (size_t)((s + 1) * F + m0) * CORRC,
                              (unsigned)(__SIZE_TYPE__)&slab[(s + 1) & 1][0], CORR_MT);
                __builtin_amdgcn_s_wait_tensorcnt(1);   // tap s complete (in-order)
            } else {
                __builtin_amdgcn_s_wait_tensorcnt(0);
            }
        }
        __syncthreads();
#else
        __syncthreads();
        {   // cooperative fallback copy of tap s
            const bf16* src = Amat + (size_t)(s * F + m0) * CORRC;
            bf16* dst = &slab[s & 1][0];
            for (int i = tid * 8; i < CORR_MT * CORRC; i += 128 * 8)
                *reinterpret_cast<v8bf*>(dst + i) =
                    *reinterpret_cast<const v8bf*>(src + i);
        }
        __syncthreads();
#endif
        const bf16* As = &slab[s & 1][0];
        const int dy = s / 7 - 3, dx = s % 7 - 3;
        const int ys = y + dy, xs = x + dx;
        const bool inb = (ys >= 0) & (ys < HH) & (xs >= 0) & (xs < HH);
        const bf16* brow = TNt + (ys * HH + xs) * CORRC;
#pragma unroll
        for (int kk = 0; kk < CORRC; kk += 32) {
            v16bf bv = bz;
            if (inb) bv = load_b_frag(brow, kk, hi);
            v16bf a0 = load_a_frag(As + (lo)      * CORRC, kk, hi);
            v16bf a1 = load_a_frag(As + (16 + lo) * CORRC, kk, hi);
            v16bf a2 = load_a_frag(As + (32 + lo) * CORRC, kk, hi);
            acc0 = __builtin_amdgcn_wmma_f32_16x16x32_bf16(false, a0, false, bv,
                                                           (short)0, acc0, false, false);
            acc1 = __builtin_amdgcn_wmma_f32_16x16x32_bf16(false, a1, false, bv,
                                                           (short)0, acc1, false, false);
            acc2 = __builtin_amdgcn_wmma_f32_16x16x32_bf16(false, a2, false, bv,
                                                           (short)0, acc2, false, false);
        }
        __syncthreads();
    }
#pragma unroll
    for (int v = 0; v < 8; ++v) {
        OUT[(m0      + v + 8 * hi) * HWPIX + n0 + lo] = acc0[v];
        OUT[(m0 + 16 + v + 8 * hi) * HWPIX + n0 + lo] = acc1[v];
        OUT[(m0 + 32 + v + 8 * hi) * HWPIX + n0 + lo] = acc2[v];
    }
}

// ---------------------------------------------------------------------------
// Target-side normalization:  e = sum_c tgt^2, nrm = sqrt(boxfilter7(e)),
// tn = 32*tgt/nrm  (stored bf16, transposed to [pixel][channel]).
// ---------------------------------------------------------------------------
__global__ void energy_kernel(const float* __restrict__ tgt_cb, float* __restrict__ e) {
    int t = blockIdx.x * 256 + threadIdx.x;   // t < 8192
    int g = t >> 12, p = t & 4095;
    float s = 0.f;
    for (int c = 0; c < CORRC; ++c) {
        float v = tgt_cb[((g << 8) + c) * HWPIX + p];
        s += v * v;
    }
    e[t] = s;
}
__global__ void scale_kernel(const float* __restrict__ e, float* __restrict__ sc) {
    int t = blockIdx.x * 256 + threadIdx.x;   // t < 8192
    int g = t >> 12, p = t & 4095;
    int y = p >> 6, x = p & 63;
    float s = 0.f;
    for (int dy = -3; dy <= 3; ++dy) {
        int ys = y + dy;
        if (ys < 0 || ys >= HH) continue;
        for (int dx = -3; dx <= 3; ++dx) {
            int xs = x + dx;
            if (xs < 0 || xs >= HH) continue;
            s += e[g * HWPIX + ys * HH + xs];
        }
    }
    sc[t] = 32.f / sqrtf(s);
}
__global__ void tn_kernel(const float* __restrict__ tgt_cb, const float* __restrict__ sc,
                          bf16* __restrict__ tnt) {
    int b = blockIdx.x;                       // g*4096 + p
    int g = b >> 12, p = b & 4095;
    float s = sc[b];
    int c = threadIdx.x;
    tnt[(long)g * HWPIX * CORRC + p * CORRC + c] =
        (bf16)(tgt_cb[((g << 8) + c) * HWPIX + p] * s);
}

// ---------------------------------------------------------------------------
// Kernel-side normalization: per filter row (CORR*49 elems) L2-normalize and
// repack to [s][f][j] bf16 for the correlation GEMMs.
// ---------------------------------------------------------------------------
__global__ void knorm_kernel(const float* __restrict__ kraw,     // [13824][392]
                             bf16* __restrict__ Acls,            // [49][144][256]
                             bf16* __restrict__ Abox) {          // [49][288][256]
    __shared__ float red[256];
    int b = blockIdx.x, j = threadIdx.x;
    int rowbase, colbase, F, f;
    bf16* dst;
    if (b < F_CLS) { int n = b / 18, a = b % 18;
        rowbase = a * CORRC; colbase = n * 49; F = F_CLS; f = b; dst = Acls; }
    else { int b2 = b - F_CLS; int n = b2 / 36, a = b2 % 36;
        rowbase = 4608 + a * CORRC; colbase = n * 49; F = F_BOX; f = b2; dst = Abox; }
    float part = 0.f;
    for (int s = 0; s < 49; ++s) {
        float v = kraw[(long)(rowbase + j) * QCOL + colbase + s];
        part += v * v;
    }
    red[j] = part; __syncthreads();
    for (int off = 128; off > 0; off >>= 1) {
        if (j < off) red[j] += red[j + off];
        __syncthreads();
    }
    float inv = 1.f / sqrtf(red[0]);
    for (int s = 0; s < 49; ++s)
        dst[((long)(s * F) + f) * CORRC + j] =
            (bf16)(kraw[(long)(rowbase + j) * QCOL + colbase + s] * inv);
}

// ---------------------------------------------------------------------------
// 36x36 adjust conv per batch, V_WMMA_F32_16X16X4_F32 (f32 matrix op).
// ---------------------------------------------------------------------------
__global__ void adjust_kernel(const float* __restrict__ boxc,    // [288][4096]
                              const float* __restrict__ Wadj,    // [36][36]
                              const float* __restrict__ badj,    // [36]
                              float* __restrict__ bbox) {        // [8][36][4096]
    const int n0 = blockIdx.x * 16;           // pixel tile
    const int m0 = blockIdx.y * 16;           // out-channel tile (0,16,32)
    const int bn = blockIdx.z;
    const int l  = threadIdx.x, lo = l & 15, hi = l >> 4;
    const float* B = boxc + (long)bn * 36 * HWPIX;
#if __has_builtin(__builtin_amdgcn_wmma_f32_16x16x4_f32)
    const int m = m0 + lo;
    v8f acc = {};
    for (int kb = 0; kb < 36; kb += 4) {
        int k0 = kb + 2 * hi;
        v2f a, b;
        a.x = (m < 36) ? Wadj[m * 36 + k0]     : 0.f;
        a.y = (m < 36) ? Wadj[m * 36 + k0 + 1] : 0.f;
        b.x = B[(k0)     * HWPIX + n0 + lo];
        b.y = B[(k0 + 1) * HWPIX + n0 + lo];
        acc = __builtin_amdgcn_wmma_f32_16x16x4_f32(false, a, false, b,
                                                    (short)0, acc, false, false);
    }
#pragma unroll
    for (int v = 0; v < 8; ++v) {
        int mg = m0 + v + 8 * hi;
        if (mg < 36)
            bbox[((long)bn * 36 + mg) * HWPIX + n0 + lo] = acc[v] + badj[mg];
    }
#else
    for (int v = 0; v < 8; ++v) {
        int mg = m0 + v + 8 * hi;
        if (mg >= 36) continue;
        float s = badj[mg];
        for (int k = 0; k < 36; ++k)
            s += Wadj[mg * 36 + k] * B[k * HWPIX + n0 + lo];
        bbox[((long)bn * 36 + mg) * HWPIX + n0 + lo] = s;
    }
#endif
}

// ---------------------------------------------------------------------------
// Softmax over fg/bg pairs + score extraction.
// ---------------------------------------------------------------------------
__global__ void softmax_kernel(const float* __restrict__ cls,    // [144][4096]
                               float* __restrict__ cp,           // [8][18][4096]
                               float* __restrict__ scores) {     // [8][36864]
    int t = blockIdx.x * 256 + threadIdx.x;   // t < 8*9*4096
    int n = t / (NANC * HWPIX);
    int r = t % (NANC * HWPIX);
    int a = r / HWPIX, p = r % HWPIX;
    float s0 = cls[(n * 18 + a) * HWPIX + p];
    float s1 = cls[(n * 18 + a + NANC) * HWPIX + p];
    float m  = fmaxf(s0, s1);
    float e0 = expf(s0 - m), e1 = expf(s1 - m);
    float inv = 1.f / (e0 + e1);
    cp[(n * 18 + a) * HWPIX + p]        = e0 * inv;
    cp[(n * 18 + a + NANC) * HWPIX + p] = e1 * inv;
    scores[(long)n * 36864 + p * NANC + a] = e1 * inv;
}

// ---------------------------------------------------------------------------
// Anchor decode + clip (anchors generated analytically, matching _ANCHORS).
// ---------------------------------------------------------------------------
__global__ void decode_kernel(const float* __restrict__ bbox,    // [8][36][4096]
                              const float* __restrict__ im_info,
                              float* __restrict__ props) {       // [8][36864][4]
    int t = blockIdx.x * 256 + threadIdx.x;   // t < 8*36864
    int n = t / 36864, i = t % 36864;
    int p = i / NANC, a = i % NANC;
    int y = p >> 6, x = p & 63;
    int rr = a / 3, si = a % 3;
    float ratio = (rr == 0) ? 0.5f : ((rr == 1) ? 1.0f : 2.0f);
    float ws0 = roundf(sqrtf(256.f / ratio));
    float hs0 = roundf(ws0 * ratio);
    float scl = (si == 0) ? 8.f : ((si == 1) ? 16.f : 32.f);
    float wa = ws0 * scl, ha = hs0 * scl;
    float acx = x * 16.f + 8.f, acy = y * 16.f + 8.f;
    const float* bb = bbox + ((long)n * 36) * HWPIX;
    float dx = bb[(4 * a + 0) * HWPIX + p];
    float dy = bb[(4 * a + 1) * HWPIX + p];
    float dw = bb[(4 * a + 2) * HWPIX + p];
    float dh = bb[(4 * a + 3) * HWPIX + p];
    float pcx = dx * wa + acx, pcy = dy * ha + acy;
    float pw = expf(dw) * wa,  ph = expf(dh) * ha;
    float imH = im_info[0] - 1.f, imW = im_info[1] - 1.f;
    float x1 = fminf(fmaxf(pcx - 0.5f * pw, 0.f), imW);
    float y1 = fminf(fmaxf(pcy - 0.5f * ph, 0.f), imH);
    float x2 = fminf(fmaxf(pcx + 0.5f * pw, 0.f), imW);
    float y2 = fminf(fmaxf(pcy + 0.5f * ph, 0.f), imH);
    float* o = props + (long)t * 4;
    o[0] = x1; o[1] = y1; o[2] = x2; o[3] = y2;
}

// ---------------------------------------------------------------------------
// Exact top-2000 per batch (iterative block argmax, values held in registers).
// ---------------------------------------------------------------------------
__global__ void topk_kernel(const float* __restrict__ scores, int* __restrict__ topi) {
    int n = blockIdx.x, tid = threadIdx.x;    // 1024 threads, 36864 = 1024*36
    const float* s = scores + (long)n * 36864;
    float v[36];
#pragma unroll
    for (int j = 0; j < 36; ++j) v[j] = s[tid * 36 + j];
    __shared__ float sv[1024];
    __shared__ int   si[1024];
    for (int t = 0; t < PRE_N; ++t) {
        float bv = NEGINF; int bi = 0;
#pragma unroll
        for (int j = 0; j < 36; ++j) {
            int gi = tid * 36 + j;
            if (v[j] > bv || (v[j] == bv && gi < bi)) { bv = v[j]; bi = gi; }
        }
        sv[tid] = bv; si[tid] = bi;
        __syncthreads();
        for (int off = 512; off > 0; off >>= 1) {
            if (tid < off) {
                float ov = sv[tid + off]; int oi = si[tid + off];
                if (ov > sv[tid] || (ov == sv[tid] && oi < si[tid])) { sv[tid] = ov; si[tid] = oi; }
            }
            __syncthreads();
        }
        int g = si[0];
        if (tid == 0) topi[n * PRE_N + t] = g;
#pragma unroll
        for (int j = 0; j < 36; ++j)
            if (tid * 36 + j == g) v[j] = NEGINF;
        __syncthreads();
    }
}
__global__ void gather_kernel(const float* __restrict__ props, const int* __restrict__ topi,
                              float* __restrict__ topb) {
    int t = blockIdx.x * 256 + threadIdx.x;
    if (t >= NTPL * PRE_N) return;
    int n = t / PRE_N;
    int idx = topi[t];
    const float* src = props + ((long)n * 36864 + idx) * 4;
    float* dst = topb + (long)t * 4;
    dst[0] = src[0]; dst[1] = src[1]; dst[2] = src[2]; dst[3] = src[3];
}

// ---------------------------------------------------------------------------
// Greedy NMS in LDS + ROI emission (zero-fills the ROI slab first).
// ---------------------------------------------------------------------------
__global__ void nms_kernel(const float* __restrict__ topb, float* __restrict__ rois) {
    __shared__ float X1[PRE_N], Y1[PRE_N], X2[PRE_N], Y2[PRE_N], AR[PRE_N];
    __shared__ int   keep[PRE_N];
    int n = blockIdx.x, tid = threadIdx.x;    // 256 threads
    const float* b = topb + (long)n * PRE_N * 4;
    for (int i = tid; i < PRE_N; i += 256) {
        float x1 = b[i * 4 + 0], y1 = b[i * 4 + 1], x2 = b[i * 4 + 2], y2 = b[i * 4 + 3];
        X1[i] = x1; Y1[i] = y1; X2[i] = x2; Y2[i] = y2;
        AR[i] = (x2 - x1 + 1.f) * (y2 - y1 + 1.f);
        keep[i] = 1;
    }
    float* r = rois + n * POST_N * 5;
    for (int i = tid; i < POST_N * 5; i += 256) r[i] = 0.f;
    __syncthreads();
    for (int i = 0; i < PRE_N - 1; ++i) {
        if (keep[i]) {
            float x1 = X1[i], y1 = Y1[i], x2 = X2[i], y2 = Y2[i], ai = AR[i];
            for (int j = i + 1 + tid; j < PRE_N; j += 256) {
                float iw = fmaxf(fminf(x2, X2[j]) - fmaxf(x1, X1[j]) + 1.f, 0.f);
                float ih = fmaxf(fminf(y2, Y2[j]) - fmaxf(y1, Y1[j]) + 1.f, 0.f);
                float inter = iw * ih;
                float iou = inter / (ai + AR[j] - inter);
                if (iou > 0.7f) keep[j] = 0;
            }
        }
        __syncthreads();
    }
    if (tid == 0) {
        int rank = 0;
        for (int i = 0; i < PRE_N; ++i) {
            if (keep[i]) {
                if (rank < POST_N) {
                    r[rank * 5 + 0] = (float)n;
                    r[rank * 5 + 1] = X1[i]; r[rank * 5 + 2] = Y1[i];
                    r[rank * 5 + 3] = X2[i]; r[rank * 5 + 4] = Y2[i];
                }
                ++rank;
            }
        }
    }
}

// ---------------------------------------------------------------------------
// Host-side orchestration.
// ---------------------------------------------------------------------------
extern "C" void kernel_launch(void* const* d_in, const int* in_sizes, int n_in,
                              void* d_out, int out_size, void* d_ws, size_t ws_size,
                              hipStream_t stream) {
    const float* tgt  = (const float*)d_in[0];
    const float* tpl  = (const float*)d_in[1];
    const float* info = (const float*)d_in[2];
    const float* Wc   = (const float*)d_in[3];
    const float* bc   = (const float*)d_in[4];
    const float* Wb   = (const float*)d_in[5];
    const float* bbx  = (const float*)d_in[6];
    const float* Wcs  = (const float*)d_in[7];
    const float* bcs  = (const float*)d_in[8];
    const float* Wbp  = (const float*)d_in[9];
    const float* bbp  = (const float*)d_in[10];
    const float* Wadj = (const float*)d_in[11];
    const float* badj = (const float*)d_in[12];

    float* out  = (float*)d_out;
    float* rois = out;                                   // 8*300*5
    float* cp   = out + 12000;                           // 8*18*4096
    float* bbox = out + 12000 + 8 * 18 * HWPIX;          // 8*36*4096

    char* ws = (char*)d_ws;
    size_t off = 0;
    auto alloc = [&](size_t bytes) { size_t r = off; off = (off + bytes + 255) & ~(size_t)255; return r; };
    bf16*  Xt      = (bf16*)(ws + alloc((size_t)HWPIX * DIN * 2));
    bf16*  Wccb    = (bf16*)(ws + alloc((size_t)512 * DIN * 2));
    bf16*  Wkb     = (bf16*)(ws + alloc((size_t)MKROW * DIN * 2));
    bf16*  Tt      = (bf16*)(ws + alloc((size_t)QCOL * DIN * 2));
    float* biascc  = (float*)(ws + alloc((size_t)512 * 4));
    float* biask   = (float*)(ws + alloc((size_t)MKROW * 4));
    float* tgt_cb  = (float*)(ws + alloc((size_t)512 * HWPIX * 4));
    float* kraw    = (float*)(ws + alloc((size_t)MKROW * QCOL * 4));
    bf16*  Acls    = (bf16*)(ws + alloc((size_t)49 * F_CLS * CORRC * 2));
    bf16*  Abox    = (bf16*)(ws + alloc((size_t)49 * F_BOX * CORRC * 2));
    float* energy  = (float*)(ws + alloc((size_t)2 * HWPIX * 4));
    float* scl     = (float*)(ws + alloc((size_t)2 * HWPIX * 4));
    bf16*  TNt     = (bf16*)(ws + alloc((size_t)2 * HWPIX * CORRC * 2));
    float* outcls  = (float*)(ws + alloc((size_t)F_CLS * HWPIX * 4));
    float* outbox  = (float*)(ws + alloc((size_t)F_BOX * HWPIX * 4));
    float* scores  = (float*)(ws + alloc((size_t)NTPL * 36864 * 4));
    float* props   = (float*)(ws + alloc((size_t)NTPL * 36864 * 4 * 4));
    int*   topi    = (int*)  (ws + alloc((size_t)NTPL * PRE_N * 4));
    float* topb    = (float*)(ws + alloc((size_t)NTPL * PRE_N * 4 * 4));

    // 1) pack operands to bf16 in WMMA-friendly layouts
    pack_bias2_kernel<<<dim3(2), dim3(256), 0, stream>>>(bc, bbx, biascc, CORRC, 512);
    pack_bias2_kernel<<<dim3(54), dim3(256), 0, stream>>>(bcs, bbp, biask, 4608, MKROW);
    pack_wcc_kernel<<<dim3(512), dim3(256), 0, stream>>>(Wc, Wb, Wccb);
    pack_wk_kernel<<<dim3(MKROW), dim3(256), 0, stream>>>(Wcs, Wbp, Wkb);
    pack_xt_kernel<<<dim3(HWPIX), dim3(256), 0, stream>>>(tgt, Xt);
    pack_tt_kernel<<<dim3(QCOL), dim3(256), 0, stream>>>(tpl, Tt);

    // 2) conv1x1 GEMMs (bf16 WMMA, 64x16 tiles per wave)
    gemm_bf16_kernel<<<dim3(HWPIX / 16, 512 / 64), dim3(32), 0, stream>>>(
        Wccb, Xt, tgt_cb, biascc, HWPIX, DIN);
    gemm_bf16_kernel<<<dim3((QCOL + 15) / 16, MKROW / 64), dim3(32), 0, stream>>>(
        Wkb, Tt, kraw, biask, QCOL, DIN);

    // 3) target normalization -> TNt (bf16 [pixel][channel])
    energy_kernel<<<dim3(32), dim3(256), 0, stream>>>(tgt_cb, energy);
    scale_kernel<<<dim3(32), dim3(256), 0, stream>>>(energy, scl);
    tn_kernel<<<dim3(2 * HWPIX), dim3(256), 0, stream>>>(tgt_cb, scl, TNt);

    // 4) kernel normalization -> [s][f][j] bf16
    knorm_kernel<<<dim3(F_CLS + F_BOX), dim3(256), 0, stream>>>(kraw, Acls, Abox);

    // 5) 7x7 cross-correlations (WMMA implicit GEMM, TDM-staged LDS A-slabs)
    corr_wmma_kernel<<<dim3(HWPIX / CORR_NT, F_CLS / CORR_MT), dim3(128), 0, stream>>>(
        Acls, TNt, outcls, F_CLS);
    corr_wmma_kernel<<<dim3(HWPIX / CORR_NT, F_BOX / CORR_MT), dim3(128), 0, stream>>>(
        Abox, TNt + (size_t)HWPIX * CORRC, outbox, F_BOX);

    // 6) 36x36 adjust conv (f32 WMMA) -> bbox_pred output slab
    adjust_kernel<<<dim3(HWPIX / 16, 3, NTPL), dim3(32), 0, stream>>>(
        outbox, Wadj, badj, bbox);

    // 7) softmax + scores, anchor decode
    softmax_kernel<<<dim3(NTPL * NANC * HWPIX / 256), dim3(256), 0, stream>>>(
        outcls, cp, scores);
    decode_kernel<<<dim3(NTPL * 36864 / 256), dim3(256), 0, stream>>>(
        bbox, info, props);

    // 8) top-2000, gather, NMS, ROI emission
    topk_kernel<<<dim3(NTPL), dim3(1024), 0, stream>>>(scores, topi);
    gather_kernel<<<dim3((NTPL * PRE_N + 255) / 256), dim3(256), 0, stream>>>(
        props, topi, topb);
    nms_kernel<<<dim3(NTPL), dim3(256), 0, stream>>>(topb, rois);

    (void)in_sizes; (void)n_in; (void)out_size; (void)ws_size;
}
